// LmToKG_35450660062011
// MI455X (gfx1250) — matrix operand order repaired and checked by reading, not compiled, not run on hardware
//
#include <hip/hip_runtime.h>
#include <hip/hip_bf16.h>

#define BB   16
#define NKG  1024
#define SLM  1024
#define DD   1024
#define HH   4096

typedef __attribute__((ext_vector_type(16))) __bf16 v16bf;
typedef __attribute__((ext_vector_type(8)))  __bf16 v8bf;
typedef __attribute__((ext_vector_type(8)))  float  v8f;

// ---------- bf16 helpers (RNE) ----------
__device__ __forceinline__ __bf16 f2bf(float f) {
    union { float f; unsigned u; } v; v.f = f;
    unsigned r = v.u + 0x7FFFu + ((v.u >> 16) & 1u);
    unsigned short h = (unsigned short)(r >> 16);
    union { unsigned short u; __bf16 b; } o; o.u = h;
    return o.b;
}
__device__ __forceinline__ float bf2f(__bf16 b) {
    union { unsigned short u; __bf16 b; } i; i.b = b;
    union { unsigned u; float f; } v; v.u = ((unsigned)i.u) << 16;
    return v.f;
}

// ---------- elementwise fp32 -> bf16 ----------
__global__ void convert_bf16_kernel(const float* __restrict__ in,
                                    __bf16* __restrict__ out, size_t n) {
    size_t stride = (size_t)gridDim.x * blockDim.x;
    for (size_t i = (size_t)blockIdx.x * blockDim.x + threadIdx.x; i < n; i += stride)
        out[i] = f2bf(in[i]);
}

// ---------- LDS-tiled transpose: in f32 [R,C] -> out bf16 [C,R], batched ----------
__global__ void transpose_bf16_kernel(const float* __restrict__ in,
                                      __bf16* __restrict__ out, int R, int C) {
    __shared__ float tile[32][33];
    size_t batchOff = (size_t)blockIdx.z * (size_t)R * C;
    const float* inp = in + batchOff;
    __bf16* outp = out + batchOff;
    int bx = blockIdx.x * 32;  // C dim
    int by = blockIdx.y * 32;  // R dim
    int tx = threadIdx.x, ty = threadIdx.y;
#pragma unroll
    for (int j = 0; j < 4; ++j)
        tile[ty + 8 * j][tx] = inp[(size_t)(by + ty + 8 * j) * C + bx + tx];
    __syncthreads();
#pragma unroll
    for (int j = 0; j < 4; ++j)
        outp[(size_t)(bx + ty + 8 * j) * R + by + tx] = f2bf(tile[tx][ty + 8 * j]);
}

// ======================================================================
// Tiled NT GEMM, double-buffered LDS fed by async global->LDS DMA.
//   C[M,N] = A[M,K] * B[N,K]^T  (bf16 in, f32 accumulate, WMMA 16x16x32)
// Block: 256 threads = 8 waves (4x2). Block tile 128x128, wave tile 32x64.
// Per K-step (32): GLOBAL_LOAD_ASYNC_TO_LDS_B128 prefetches the next A/B
// 128x32 tiles into the alternate LDS buffer while WMMA consumes the
// current one; sync = s_wait_asynccnt 0 + one workgroup barrier.
// ======================================================================
#define LDS_STRIDE 48                 // elements: 32 used + 16 pad (32B-aligned rows)
#define TILE_ELTS  (128 * LDS_STRIDE) // 6144 elements
#define TILE_BYTES (TILE_ELTS * 2)    // 12288 bytes per tile
// LDS byte layout inside the single SMEM object (object base == 0):
//   A0 @ 0, B0 @ 12288, A1 @ 24576, B1 @ 36864   (total 48 KB)

__device__ __forceinline__ void async_tile_cp(const __bf16* __restrict__ gbase,
                                              int ldg, int k0, unsigned ldsBase) {
    const int t = threadIdx.x;
#pragma unroll
    for (int c = 0; c < 2; ++c) {
        int chunk = t + c * 256;            // 0..511 : 128 rows x 4 16B-parts
        int row = chunk >> 2;
        int part = chunk & 3;
        const __bf16* g = gbase + (size_t)row * ldg + k0 + part * 8;
        unsigned l = ldsBase + (unsigned)(row * (LDS_STRIDE * 2) + part * 16);
        asm volatile("global_load_async_to_lds_b128 %0, %1, off"
                     :: "v"(l), "v"((unsigned long long)(uintptr_t)g)
                     : "memory");
    }
}

// A-fragment (16x32): lane(m=l&15, hi=l>>4) holds K = hi*16+0..15
__device__ __forceinline__ v16bf lds_frag_a(const __bf16* s, int row0) {
    int lane = threadIdx.x & 31;
    int m = lane & 15, hi = lane >> 4;
    const __bf16* p = s + (row0 + m) * LDS_STRIDE + hi * 16;
    v8bf lo = *(const v8bf*)p;
    v8bf hb = *(const v8bf*)(p + 8);
    v16bf r;
#pragma unroll
    for (int i = 0; i < 8; ++i) { r[i] = lo[i]; r[i + 8] = hb[i]; }
    return r;
}
// B-fragment (32x16, NT): lane(n=l&15, hi=l>>4) holds K = hi*16+0..15
__device__ __forceinline__ v16bf lds_frag_b(const __bf16* s, int col0) {
    int lane = threadIdx.x & 31;
    int n = lane & 15, hi = lane >> 4;
    return *(const v16bf*)(s + (col0 + n) * LDS_STRIDE + hi * 16);
}

// MODE 0: logits : bf16 store of (acc + logf(mask[z*SLM+col]+1e-45f))
// MODE 1: out1   : f32  store of (kg[z,row,col] + acc - c[z,col])
// MODE 2: ffn-h  : bf16 store of relu(acc + b1[col])
// MODE 3: final  : f32  store of (out1[row,col] + acc + b2[col])
template <int MODE>
__global__ __launch_bounds__(256)
void gemm_nt_kernel(const __bf16* __restrict__ Ab,
                    const __bf16* __restrict__ Bb,
                    int K, int lda, int ldb, int ldc,
                    long strideA, long strideB, long strideC,
                    void* __restrict__ Cout,
                    const float* __restrict__ ep0, long e0stride,
                    const float* __restrict__ ep1, long e1stride) {
    __shared__ __attribute__((aligned(32))) __bf16 SMEM[4 * TILE_ELTS];  // A0 B0 A1 B1

    const int z = blockIdx.z;
    const int t = threadIdx.x;
    const int wave = t >> 5;
    const int wm = wave >> 1;            // 0..3 -> 32-row slab
    const int wn = wave & 1;             // 0..1 -> 64-col slab

    const __bf16* gA = Ab + (size_t)z * strideA + (size_t)(blockIdx.y * 128) * lda;
    const __bf16* gB = Bb + (size_t)z * strideB + (size_t)(blockIdx.x * 128) * ldb;

    v8f acc[2][4];
#pragma unroll
    for (int i = 0; i < 2; ++i)
#pragma unroll
        for (int j = 0; j < 4; ++j) acc[i][j] = (v8f)(0.0f);

    const int steps = K >> 5;
    // prime buffer 0
    async_tile_cp(gA, lda, 0, 0u);
    async_tile_cp(gB, ldb, 0, (unsigned)TILE_BYTES);

    for (int kI = 0; kI < steps; ++kI) {
        const int cur = kI & 1;
        asm volatile("s_wait_asynccnt 0x0" ::: "memory");  // my tile landed in LDS
        __syncthreads();                                   // everyone's landed / prior reads done
        if (kI + 1 < steps) {                              // prefetch next into alternate buffer
            unsigned nxtByte = (unsigned)(cur ^ 1) * (2u * TILE_BYTES);
            async_tile_cp(gA, lda, (kI + 1) * 32, nxtByte);
            async_tile_cp(gB, ldb, (kI + 1) * 32, nxtByte + (unsigned)TILE_BYTES);
        }

        const __bf16* sAc = SMEM + cur * (2 * TILE_ELTS);
        const __bf16* sBc = SMEM + TILE_ELTS + cur * (2 * TILE_ELTS);

        v16bf af[2], bfr[4];
#pragma unroll
        for (int i = 0; i < 2; ++i) af[i]  = lds_frag_a(sAc, wm * 32 + 16 * i);
#pragma unroll
        for (int j = 0; j < 4; ++j) bfr[j] = lds_frag_b(sBc, wn * 64 + 16 * j);
#pragma unroll
        for (int i = 0; i < 2; ++i)
#pragma unroll
            for (int j = 0; j < 4; ++j)
                acc[i][j] = __builtin_amdgcn_wmma_f32_16x16x32_bf16(
                    false, af[i], false, bfr[j], (short)0, acc[i][j], false, false);
    }

    // epilogue: C/D layout — VGPR r at lane l: row = r + 8*(l>>4), col = l&15
    const int lane = t & 31;
    const int colL = lane & 15, hi = lane >> 4;
    const int mBase = blockIdx.y * 128 + wm * 32;
    const int nBase = blockIdx.x * 128 + wn * 64;
#pragma unroll
    for (int i = 0; i < 2; ++i) {
#pragma unroll
        for (int j = 0; j < 4; ++j) {
#pragma unroll
            for (int r = 0; r < 8; ++r) {
                int row = mBase + 16 * i + r + 8 * hi;
                int col = nBase + 16 * j + colL;
                float v = acc[i][j][r];
                size_t cidx = (size_t)z * strideC + (size_t)row * ldc + col;
                if constexpr (MODE == 0) {
                    float mask = ep0[(size_t)z * e0stride + col];
                    ((__bf16*)Cout)[cidx] = f2bf(v + logf(mask + 1e-45f));
                } else if constexpr (MODE == 1) {
                    float kg = ep0[(size_t)z * e0stride + (size_t)row * ldc + col];
                    float cc = ep1[(size_t)z * e1stride + col];
                    ((float*)Cout)[cidx] = kg + v - cc;
                } else if constexpr (MODE == 2) {
                    float h = v + ep0[col];
                    ((__bf16*)Cout)[cidx] = f2bf(h > 0.0f ? h : 0.0f);
                } else {
                    float o1 = ep0[(size_t)row * ldc + col];
                    ((float*)Cout)[cidx] = o1 + v + ep1[col];
                }
            }
        }
    }
}

// ---------- column logsumexp over NKG axis (per (b,s)), online ----------
__global__ void lse_kernel(const __bf16* __restrict__ logits,
                           float* __restrict__ lse) {
    int idx = blockIdx.x * blockDim.x + threadIdx.x;   // 0 .. B*SLM-1
    int b = idx / SLM, s = idx % SLM;
    const __bf16* p = logits + (size_t)b * NKG * SLM + s;
    float m = -3.4e38f, sum = 0.0f;
    for (int n = 0; n < NKG; ++n) {
        float x = bf2f(p[(size_t)n * SLM]);
        if (x > m) { sum = sum * __expf(m - x) + 1.0f; m = x; }
        else       { sum += __expf(x - m); }
    }
    lse[idx] = m + logf(sum);
}

// ---------- correction c[b,d] = sum_s lse[b,s] * lm[b,s,d] ----------
__global__ void corr_kernel(const float* __restrict__ lse,
                            const float* __restrict__ lm,
                            float* __restrict__ cbuf) {
    int idx = blockIdx.x * blockDim.x + threadIdx.x;   // 0 .. B*DD-1
    int b = idx / DD, d = idx % DD;
    const float* lmp = lm + (size_t)b * SLM * DD + d;
    const float* lsep = lse + (size_t)b * SLM;
    float sum = 0.0f;
    for (int s = 0; s < SLM; ++s)
        sum += lsep[s] * lmp[(size_t)s * DD];
    cbuf[idx] = sum;
}

// ---------- LayerNorm over D=1024, one row per 256-thread block ----------
__global__ void layernorm_kernel(const float* __restrict__ x,
                                 const float* __restrict__ gamma,
                                 const float* __restrict__ beta,
                                 __bf16* __restrict__ out) {
    __shared__ float r1[8], r2[8];
    size_t row = blockIdx.x;
    const float* xp = x + row * DD;
    float vals[4], s1 = 0.0f, s2 = 0.0f;
#pragma unroll
    for (int k = 0; k < 4; ++k) {
        float v = xp[threadIdx.x + 256 * k];
        vals[k] = v; s1 += v; s2 += v * v;
    }
#pragma unroll
    for (int o = 16; o > 0; o >>= 1) {
        s1 += __shfl_xor(s1, o, 32);
        s2 += __shfl_xor(s2, o, 32);
    }
    if ((threadIdx.x & 31) == 0) { r1[threadIdx.x >> 5] = s1; r2[threadIdx.x >> 5] = s2; }
    __syncthreads();
    float t1 = 0.0f, t2 = 0.0f;
#pragma unroll
    for (int i = 0; i < 8; ++i) { t1 += r1[i]; t2 += r2[i]; }
    float mu = t1 * (1.0f / DD);
    float var = t2 * (1.0f / DD) - mu * mu;
    float rs = rsqrtf(var + 1e-6f);
    __bf16* op = out + row * DD;
#pragma unroll
    for (int k = 0; k < 4; ++k) {
        int c = threadIdx.x + 256 * k;
        op[c] = f2bf((vals[k] - mu) * rs * gamma[c] + beta[c]);
    }
}

extern "C" void kernel_launch(void* const* d_in, const int* in_sizes, int n_in,
                              void* d_out, int out_size, void* d_ws, size_t ws_size,
                              hipStream_t stream) {
    const float* kg    = (const float*)d_in[0];
    const float* lm    = (const float*)d_in[1];
    const float* mask  = (const float*)d_in[2];
    const float* W1    = (const float*)d_in[3];
    const float* b1    = (const float*)d_in[4];
    const float* W2    = (const float*)d_in[5];
    const float* b2    = (const float*)d_in[6];
    const float* gamma = (const float*)d_in[7];
    const float* beta  = (const float*)d_in[8];

    // workspace layout (bytes), all offsets 256B-aligned
    char* ws = (char*)d_ws;
    size_t off = 0;
    auto alloc = [&](size_t bytes) { char* p = ws + off; off += (bytes + 255) & ~size_t(255); return p; };
    __bf16* kgb     = (__bf16*)alloc((size_t)BB * NKG * DD * 2);   // 32 MB
    __bf16* lmb     = (__bf16*)alloc((size_t)BB * SLM * DD * 2);   // 32 MB
    __bf16* lmTb    = (__bf16*)alloc((size_t)BB * DD * SLM * 2);   // 32 MB
    __bf16* w1t     = (__bf16*)alloc((size_t)HH * DD * 2);         // 8 MB
    __bf16* w2t     = (__bf16*)alloc((size_t)DD * HH * 2);         // 8 MB
    __bf16* logitsb = (__bf16*)alloc((size_t)BB * NKG * SLM * 2);  // 32 MB
    float*  lse     = (float*) alloc((size_t)BB * SLM * 4);        // 64 KB
    float*  cbuf    = (float*) alloc((size_t)BB * DD * 4);         // 64 KB
    float*  out1    = (float*) alloc((size_t)BB * NKG * DD * 4);   // 64 MB
    __bf16* xnorm   = (__bf16*)alloc((size_t)BB * NKG * DD * 2);   // 32 MB
    __bf16* hbuf    = (__bf16*)alloc((size_t)BB * NKG * HH * 2);   // 128 MB

    // 1) fp32 -> bf16 conversions
    convert_bf16_kernel<<<8192, 256, 0, stream>>>(kg, kgb, (size_t)BB * NKG * DD);
    convert_bf16_kernel<<<8192, 256, 0, stream>>>(lm, lmb, (size_t)BB * SLM * DD);
    // 2) transposed bf16 copies (for NT-form GEMMs)
    transpose_bf16_kernel<<<dim3(DD / 32, SLM / 32, BB), dim3(32, 8), 0, stream>>>(lm, lmTb, SLM, DD);
    transpose_bf16_kernel<<<dim3(HH / 32, DD / 32, 1),  dim3(32, 8), 0, stream>>>(W1, w1t, DD, HH);
    transpose_bf16_kernel<<<dim3(DD / 32, HH / 32, 1),  dim3(32, 8), 0, stream>>>(W2, w2t, HH, DD);

    // 3) GEMM1: logits[b,n,s] = kg[b,n,:]·lm[b,s,:] + log(mask+eps)  (bf16 out)
    gemm_nt_kernel<0><<<dim3(SLM / 128, NKG / 128, BB), 256, 0, stream>>>(
        kgb, lmb, DD, DD, DD, SLM,
        (long)NKG * DD, (long)SLM * DD, (long)NKG * SLM,
        (void*)logitsb, mask, (long)SLM, nullptr, 0);

    // 4) column logsumexp over NKG; correction c[b,d]
    lse_kernel<<<(BB * SLM) / 256, 256, 0, stream>>>(logitsb, lse);
    corr_kernel<<<(BB * DD) / 256, 256, 0, stream>>>(lse, lm, cbuf);

    // 5) GEMM2: out1[b,n,d] = kg + logits·lm - c   (f32 out)
    gemm_nt_kernel<1><<<dim3(DD / 128, NKG / 128, BB), 256, 0, stream>>>(
        logitsb, lmTb, SLM, SLM, SLM, DD,
        (long)NKG * SLM, (long)DD * SLM, (long)NKG * DD,
        (void*)out1, kg, (long)NKG * DD, cbuf, (long)DD);

    // 6) LayerNorm -> bf16
    layernorm_kernel<<<BB * NKG, 256, 0, stream>>>(out1, gamma, beta, xnorm);

    // 7) GEMM3: h = relu(xnorm @ W1 + b1)   (bf16 out, flat M = B*NKG)
    gemm_nt_kernel<2><<<dim3(HH / 128, (BB * NKG) / 128, 1), 256, 0, stream>>>(
        xnorm, w1t, DD, DD, DD, HH,
        0L, 0L, 0L,
        (void*)hbuf, b1, 0L, nullptr, 0);

    // 8) GEMM4: out = out1 + h @ W2 + b2   (f32 -> d_out)
    gemm_nt_kernel<3><<<dim3(DD / 128, (BB * NKG) / 128, 1), 256, 0, stream>>>(
        hbuf, w2t, HH, HH, HH, DD,
        0L, 0L, 0L,
        d_out, out1, 0L, b2, 0L);
}
